// GraphAttention_v2_10084583211261
// MI455X (gfx1250) — matrix operand order repaired and checked by reading, not compiled, not run on hardware
//
#include <hip/hip_runtime.h>

#define NN   8192            // nodes
#define FD   256             // Wh feature dim (all layers)
#define NGR  256             // graphs
#define VSTR 40              // LDS row stride (halves): 32 data + 8 pad (TDM pad 4 DW per 16 DW)

typedef __attribute__((ext_vector_type(16))) _Float16     v16h;
typedef __attribute__((ext_vector_type(8)))  _Float16     v8h;
typedef __attribute__((ext_vector_type(8)))  float        v8f;
typedef __attribute__((ext_vector_type(4)))  unsigned int u32x4;
typedef __attribute__((ext_vector_type(8)))  int          i32x8;
typedef __attribute__((ext_vector_type(4)))  int          i32x4;

// ---------------------------------------------------------------- adj -> bitmask
__global__ __launch_bounds__(256) void k_mask(const int* __restrict__ adj,
                                              unsigned int* __restrict__ mask) {
  int w = blockIdx.x * 256 + threadIdx.x;          // word index over N*N/32
  const int* p = adj + (size_t)w * 32;
  unsigned int bits = 0;
#pragma unroll
  for (int t = 0; t < 32; t++) bits |= (p[t] > 0 ? 1u : 0u) << t;
  mask[w] = bits;
}

// ------------------------------------------- weight prep: transpose + tile-fold, fp32->f16
__global__ __launch_bounds__(256) void k_prepW(const float* __restrict__ W0,
                                               const float* __restrict__ W1,
                                               const float* __restrict__ W2,
                                               _Float16* __restrict__ W0T,
                                               _Float16* __restrict__ W1fT,
                                               _Float16* __restrict__ W2fT) {
  int c = blockIdx.x;       // output feature 0..255
  int k = threadIdx.x;      // input feature 0..255
  if (k < 128) W0T[c * 128 + k] = (_Float16)W0[k * 256 + c];
  float f1 = W1[k * 256 + c] + W1[(k + 256) * 256 + c] +
             W1[(k + 512) * 256 + c] + W1[(k + 768) * 256 + c];
  W1fT[c * 256 + k] = (_Float16)f1;
  float f2 = W2[k * 256 + c] + W2[(k + 256) * 256 + c] +
             W2[(k + 512) * 256 + c] + W2[(k + 768) * 256 + c];
  W2fT[c * 256 + k] = (_Float16)f2;
}

// ------------------------------------------------ dense GEMM: Wh^T = (relu?)(A) @ B, f16 out
__global__ __launch_bounds__(256) void k_gemm(const float* __restrict__ A,
                                              const _Float16* __restrict__ BT,
                                              _Float16* __restrict__ CT,
                                              int K, int doRelu) {
  const int lane = threadIdx.x & 31;
  const int r  = lane & 15;
  const int hi = lane >> 4;
  const int wid = blockIdx.x * 8 + (threadIdx.x >> 5);
  const int tn = wid & 15;          // 16 col tiles (256/16)
  const int tm = wid >> 4;          // 512 row tiles (8192/16)
  const float*    Arow = A  + (size_t)(tm * 16 + r) * K;
  const _Float16* Bcol = BT + (size_t)(tn * 16 + r) * K;
  v8f acc = {};
  for (int kc = 0; kc < K; kc += 32) {
    float4 a0 = *(const float4*)(Arow + kc + hi * 16);
    float4 a1 = *(const float4*)(Arow + kc + hi * 16 + 4);
    float4 a2 = *(const float4*)(Arow + kc + hi * 16 + 8);
    float4 a3 = *(const float4*)(Arow + kc + hi * 16 + 12);
    if (doRelu) {
      a0.x=fmaxf(a0.x,0.f); a0.y=fmaxf(a0.y,0.f); a0.z=fmaxf(a0.z,0.f); a0.w=fmaxf(a0.w,0.f);
      a1.x=fmaxf(a1.x,0.f); a1.y=fmaxf(a1.y,0.f); a1.z=fmaxf(a1.z,0.f); a1.w=fmaxf(a1.w,0.f);
      a2.x=fmaxf(a2.x,0.f); a2.y=fmaxf(a2.y,0.f); a2.z=fmaxf(a2.z,0.f); a2.w=fmaxf(a2.w,0.f);
      a3.x=fmaxf(a3.x,0.f); a3.y=fmaxf(a3.y,0.f); a3.z=fmaxf(a3.z,0.f); a3.w=fmaxf(a3.w,0.f);
    }
    v16h af;   // A layout: elem e -> K=kc+hi*16+e
    af[0]=(_Float16)a0.x;  af[1]=(_Float16)a0.y;  af[2]=(_Float16)a0.z;  af[3]=(_Float16)a0.w;
    af[4]=(_Float16)a1.x;  af[5]=(_Float16)a1.y;  af[6]=(_Float16)a1.z;  af[7]=(_Float16)a1.w;
    af[8]=(_Float16)a2.x;  af[9]=(_Float16)a2.y;  af[10]=(_Float16)a2.z; af[11]=(_Float16)a2.w;
    af[12]=(_Float16)a3.x; af[13]=(_Float16)a3.y; af[14]=(_Float16)a3.z; af[15]=(_Float16)a3.w;
    v16h bf = *(const v16h*)(Bcol + kc + hi * 16);
    acc = __builtin_amdgcn_wmma_f32_16x16x32_f16(false, af, false, bf, (short)0, acc, false, false);
  }
#pragma unroll
  for (int i = 0; i < 8; i++)       // C layout: M = i + 8*hi, col = r
    CT[(size_t)(tn * 16 + r) * NN + tm * 16 + 8 * hi + i] = (_Float16)acc[i];
}

// ---------------------------------------------- s = Wh @ a_src, d = Wh @ a_dst (from Wh^T f16)
__global__ __launch_bounds__(256) void k_sd(const _Float16* __restrict__ WhT,
                                            const float* __restrict__ a,
                                            float* __restrict__ s, float* __restrict__ d) {
  int i = blockIdx.x * 256 + threadIdx.x;
  float ss = 0.f, dd = 0.f;
  for (int c = 0; c < FD; c++) {
    float w = (float)WhT[(size_t)c * NN + i];   // coalesced across i
    ss += w * a[c];
    dd += w * a[FD + c];
  }
  s[i] = ss;
  d[i] = dd;
}

// --------------------------------------------------------------- TDM: stage one V chunk
// 2D tile: tile_dim0=32 (contiguous j), tile_dim1=256 (feature rows), stride=NN elems, 2B elems.
// pad_enable: after every 16 DWORDs (64B row) skip 4 DWORDs (16B) -> LDS row stride 80B = VSTR.
__device__ inline void tdm_load_chunk(const _Float16* WhT, int jc, _Float16* buf) {
  unsigned long long ga  = (unsigned long long)(const void*)(WhT + jc);
  unsigned           lds = (unsigned)(unsigned long long)(void*)buf;
  u32x4 g0;
  g0[0] = 1u;                                               // count=1, user descriptor
  g0[1] = lds;                                              // lds_addr
  g0[2] = (unsigned)(ga & 0xffffffffu);                     // global_addr[31:0]
  g0[3] = (unsigned)((ga >> 32) & 0x01ffffffu) | (2u << 30);// global_addr[56:32] | type=2
  i32x8 g1;
  g1[0] = (int)((1u << 16) | (1u << 20) | (3u << 22) | (3u << 25));
          // data_size=2B | pad_enable | pad_interval=16DW | pad_amount=4DW
  g1[1] = (int)(8192u << 16);   // tensor_dim0 = 8192 (bits 79:48, low 16 here)
  g1[2] = (int)(256u  << 16);   // tensor_dim0 hi=0 ; tensor_dim1 = 256 (low 16)
  g1[3] = (int)(32u   << 16);   // tensor_dim1 hi=0 ; tile_dim0 = 32
  g1[4] = 256;                  // tile_dim1 = 256 ; tile_dim2 = 0
  g1[5] = 8192;                 // tensor_dim0_stride[31:0] (data_size units)
  g1[6] = 0;                    // stride hi, tensor_dim1_stride lo
  g1[7] = 0;
  i32x4 z4 = {0, 0, 0, 0};
  i32x8 z8 = {0, 0, 0, 0, 0, 0, 0, 0};
  __builtin_amdgcn_tensor_load_to_lds(g0, g1, z4, z4, z8, 0);   // 6-arg toolchain form
}

// -------------------------------------------------- flash masked-softmax attention @ Wh
// Per wave: 16 rows x 128 cols; two col-half waves per 16-row group; 64 rows/block.
// V chunks staged by TDM, double-buffered, DMA overlapped with exp+WMMA compute.
__global__ __launch_bounds__(256) void k_flash(const _Float16* __restrict__ WhT,
                                               const float* __restrict__ s,
                                               const float* __restrict__ dv,
                                               const unsigned int* __restrict__ mask,
                                               float* __restrict__ o) {
  __shared__ _Float16 Vlds[2][256 * VSTR];
  const int tid  = threadIdx.x;
  const int lane = tid & 31;
  const int wave = tid >> 5;
  const int r  = lane & 15;
  const int hi = lane >> 4;
  const int rowBase = blockIdx.x * 64 + (wave >> 1) * 16;
  const int colBase = (wave & 1) * 128;
  const int row = rowBase + r;
  const float sr = s[row];
  const unsigned int* mwrow = mask + (size_t)row * (NN / 32);
  const float4* d4 = (const float4*)dv;

  // kick off chunk 0 DMA; it overlaps the entire pass-1 scan below
  if (wave == 0) tdm_load_chunk(WhT, 0, &Vlds[0][0]);

  // ---- pass 1: exact masked row max of leaky_relu(s_i + d_j) (monotone -> masked max)
  float m = -1e9f;
  for (int w = hi; w < NN / 32; w += 2) {
    unsigned int mb = mwrow[w];
#pragma unroll
    for (int q = 0; q < 8; q++) {
      float4 da = d4[w * 8 + q];
#define UPD1(dval, t) { float v_ = sr + (dval); v_ = fmaxf(v_, 0.2f * v_); \
                        m = fmaxf(m, ((mb >> (t)) & 1u) ? v_ : -1e9f); }
      UPD1(da.x, 4 * q) UPD1(da.y, 4 * q + 1) UPD1(da.z, 4 * q + 2) UPD1(da.w, 4 * q + 3)
#undef UPD1
    }
  }
  m = fmaxf(m, __shfl_xor(m, 16, 32));

  // ---- pass 2: p = exp(e - m), sum, and P @ V via WMMA (TDM double-buffered V)
  const unsigned char* mrow = (const unsigned char*)mask + (size_t)row * (NN / 8);
  float sum = 0.f;
  v8f acc[8] = {};
  for (int i = 0; i < NN / 32; i++) {
    const int jc = i * 32;
    if (wave == 0) {
      if (jc + 32 < NN) {   // prefetch next chunk into the other buffer via TDM
        tdm_load_chunk(WhT, jc + 32, &Vlds[(i + 1) & 1][0]);
        __builtin_amdgcn_s_wait_tensorcnt((short)1);  // in-order: oldest (chunk i) landed
      } else {
        __builtin_amdgcn_s_wait_tensorcnt((short)0);
      }
    }
    __syncthreads();                       // publish chunk i to all waves
    const _Float16* Vbuf = &Vlds[i & 1][0];

    const int j0 = jc + hi * 8;
    const int j1 = jc + 16 + hi * 8;
    const unsigned int mb0 = mrow[j0 >> 3];
    const unsigned int mb1 = mrow[j1 >> 3];
    float4 d0a = d4[(j0 >> 2)], d0b = d4[(j0 >> 2) + 1];
    float4 d1a = d4[(j1 >> 2)], d1b = d4[(j1 >> 2) + 1];
    v16h pf;
#define PSET(idx, dval, mb, t) { float v_ = sr + (dval); v_ = fmaxf(v_, 0.2f * v_); \
                                 v_ = ((mb >> (t)) & 1u) ? v_ : -1e9f;              \
                                 float p_ = __expf(v_ - m); sum += p_;              \
                                 pf[idx] = (_Float16)p_; }
    PSET(0,  d0a.x, mb0, 0) PSET(1,  d0a.y, mb0, 1) PSET(2,  d0a.z, mb0, 2) PSET(3,  d0a.w, mb0, 3)
    PSET(4,  d0b.x, mb0, 4) PSET(5,  d0b.y, mb0, 5) PSET(6,  d0b.z, mb0, 6) PSET(7,  d0b.w, mb0, 7)
    PSET(8,  d1a.x, mb1, 0) PSET(9,  d1a.y, mb1, 1) PSET(10, d1a.z, mb1, 2) PSET(11, d1a.w, mb1, 3)
    PSET(12, d1b.x, mb1, 4) PSET(13, d1b.y, mb1, 5) PSET(14, d1b.z, mb1, 6) PSET(15, d1b.w, mb1, 7)
#undef PSET

#pragma unroll
    for (int ct = 0; ct < 8; ct++) {
      const _Float16* bp = &Vbuf[(colBase + ct * 16 + r) * VSTR + hi * 8];
      v8h b0 = *(const v8h*)bp;
      v8h b1 = *(const v8h*)(bp + 16);
      v16h bf = __builtin_shufflevector(b0, b1, 0,1,2,3,4,5,6,7,8,9,10,11,12,13,14,15);
      acc[ct] = __builtin_amdgcn_wmma_f32_16x16x32_f16(false, pf, false, bf, (short)0,
                                                       acc[ct], false, false);
    }
    __syncthreads();   // all reads of buf[i&1] done before it is DMA-overwritten at i+1
  }
  sum += __shfl_xor(sum, 16, 32);

  float inv[8];
#pragma unroll
  for (int i = 0; i < 8; i++) inv[i] = 1.f / __shfl(sum, 8 * hi + i, 32);

#pragma unroll
  for (int ct = 0; ct < 8; ct++)
#pragma unroll
    for (int i = 0; i < 8; i++)
      o[(size_t)(rowBase + 8 * hi + i) * FD + colBase + ct * 16 + r] = acc[ct][i] * inv[i];
}

// ------------------------------------------------------------ readout: segment mean
__global__ __launch_bounds__(256) void k_zero(float* sums, int* cnts) {
  int i = blockIdx.x * 256 + threadIdx.x;
  sums[i] = 0.f;
  if (i < NGR) cnts[i] = 0;
}
__global__ __launch_bounds__(256) void k_count(const long long* __restrict__ batch, int* cnts) {
  int i = blockIdx.x * 256 + threadIdx.x;
  atomicAdd(&cnts[(int)batch[i]], 1);
}
__global__ __launch_bounds__(256) void k_accum(const float* __restrict__ o,
                                               const long long* __restrict__ batch,
                                               float* __restrict__ sums) {
  int i = blockIdx.x, c = threadIdx.x;
  int g = (int)batch[i];
  atomicAdd(&sums[g * FD + c], fmaxf(o[(size_t)i * FD + c], 0.f));  // final relu fused here
}
__global__ __launch_bounds__(256) void k_final(const float* __restrict__ sums,
                                               const int* __restrict__ cnts,
                                               float* __restrict__ out) {
  int g = blockIdx.x, c = threadIdx.x;
  out[g * FD + c] = sums[g * FD + c] / fmaxf((float)cnts[g], 1.f);
}

// ----------------------------------------------------------------------------------
extern "C" void kernel_launch(void* const* d_in, const int* in_sizes, int n_in,
                              void* d_out, int out_size, void* d_ws, size_t ws_size,
                              hipStream_t stream) {
  (void)in_sizes; (void)n_in; (void)out_size; (void)ws_size;
  const float*     x    = (const float*)d_in[0];
  const int*       adj  = (const int*)d_in[1];
  const long long* batch= (const long long*)d_in[2];
  const float*     W0   = (const float*)d_in[3];
  const float*     a0   = (const float*)d_in[4];
  const float*     W1   = (const float*)d_in[5];
  const float*     a1   = (const float*)d_in[6];
  const float*     W2   = (const float*)d_in[7];
  const float*     a2   = (const float*)d_in[8];

  char* p = (char*)d_ws;
  auto nxt = [&](size_t b) { void* q = (void*)p; p += (b + 255) & ~(size_t)255; return q; };
  unsigned int* mask = (unsigned int*)nxt((size_t)NN * NN / 8);   // 8 MB
  _Float16*     WhT  = (_Float16*)nxt((size_t)FD * NN * 2);       // 4 MB
  float*        sv   = (float*)nxt((size_t)NN * 4);
  float*        dv   = (float*)nxt((size_t)NN * 4);
  float*        o    = (float*)nxt((size_t)NN * FD * 4);          // 8 MB
  _Float16*     W0T  = (_Float16*)nxt((size_t)256 * 128 * 2);
  _Float16*     W1fT = (_Float16*)nxt((size_t)256 * 256 * 2);
  _Float16*     W2fT = (_Float16*)nxt((size_t)256 * 256 * 2);
  float*        sums = (float*)nxt((size_t)NGR * FD * 4);
  int*          cnts = (int*)nxt((size_t)NGR * 4);

  k_mask <<<NN * NN / 32 / 256, 256, 0, stream>>>(adj, mask);
  k_prepW<<<256, 256, 0, stream>>>(W0, W1, W2, W0T, W1fT, W2fT);

  // layer 0
  k_gemm <<<1024, 256, 0, stream>>>(x, W0T, WhT, 128, 0);
  k_sd   <<<NN / 256, 256, 0, stream>>>(WhT, a0, sv, dv);
  k_flash<<<128, 256, 0, stream>>>(WhT, sv, dv, mask, o);
  // layer 1 (tile folded into W1fT, relu on A in k_gemm)
  k_gemm <<<1024, 256, 0, stream>>>(o, W1fT, WhT, 256, 1);
  k_sd   <<<NN / 256, 256, 0, stream>>>(WhT, a1, sv, dv);
  k_flash<<<128, 256, 0, stream>>>(WhT, sv, dv, mask, o);
  // layer 2
  k_gemm <<<1024, 256, 0, stream>>>(o, W2fT, WhT, 256, 1);
  k_sd   <<<NN / 256, 256, 0, stream>>>(WhT, a2, sv, dv);
  k_flash<<<128, 256, 0, stream>>>(WhT, sv, dv, mask, o);

  // segment mean readout
  k_zero <<<NGR * FD / 256, 256, 0, stream>>>(sums, cnts);
  k_count<<<NN / 256, 256, 0, stream>>>(batch, cnts);
  k_accum<<<NN, 256, 0, stream>>>(o, batch, sums);
  k_final<<<NGR, 256, 0, stream>>>(sums, cnts, (float*)d_out);
}
